// VisionTransformer_5686536699943
// MI455X (gfx1250) — matrix operand order repaired and checked by reading, not compiled, not run on hardware
//
#include <hip/hip_runtime.h>
#include <hip/hip_bf16.h>

// ---------------------------------------------------------------------------
// ViT-Base forward (BATCH=32, 197 tokens, 12 layers) for gfx1250 (MI455X).
// All matmuls via V_WMMA_F32_16X16X32_F16 (f16 inputs, f32 accumulate).
// Double-buffered LDS tiles filled with GLOBAL_LOAD_ASYNC_TO_LDS_B128
// (ASYNCcnt), one barrier per K-step.
// ---------------------------------------------------------------------------

#define USE_ASYNC 1

#define BATCH   32
#define NTOK    197
#define NPAD    224          // 7*32: multiple of 32 so it can be a WMMA K dim
#define DIM     768
#define HEADS   12
#define HD      64
#define FF      3072
#define DEPTH   12
#define GRID_P  14
#define NPATCH  196
#define MTOK    (BATCH * NTOK)      // 6304 (multiple of 16)
#define MPATCH  (BATCH * NPATCH)   // 6272
#define ZHEADS  (BATCH * HEADS)    // 384
#define NRD     732

typedef __attribute__((ext_vector_type(16))) _Float16 v16h;
typedef __attribute__((ext_vector_type(8)))  _Float16 v8h;
typedef __attribute__((ext_vector_type(8)))  float    v8f;

#define BM  64
#define BN  256
#define BK  32
#define LDT 40     // BK + 8 halves pad (avoid LDS bank conflicts)

// ---------------------------------------------------------------------------
// Fragment loaders following CDNA5 ISA 7.12.2 VGPR layouts (wave32).
// A (16x32 f16): lane m=lane&15, g=lane>>4; halves are two 16B chunks at
//   k = 8g..8g+7 (VGPR0-3) and k = 16+8g..16+8g+7 (VGPR4-7).
// B (32x16 f16): lane n=lane&15, g=lane>>4; same K chunks as A (8g.., 16+8g..) of row n.
// C/D (16x16 f32): VGPR r holds M = r + 8*(lane>>4), N = lane&15.
// ---------------------------------------------------------------------------
__device__ __forceinline__ v16h fragA(const _Float16* T, int mOff, int lane) {
  const _Float16* row = T + (size_t)(mOff + (lane & 15)) * LDT;
  int g = lane >> 4;
  v8h lo = *(const v8h*)(row + 8 * g);
  v8h hi = *(const v8h*)(row + 16 + 8 * g);
  v16h a;
#pragma unroll
  for (int i = 0; i < 8; ++i) { a[i] = lo[i]; a[8 + i] = hi[i]; }
  return a;
}

__device__ __forceinline__ v16h fragB(const _Float16* T, int nOff, int lane) {
  const _Float16* row = T + (size_t)(nOff + (lane & 15)) * LDT;
  int g = lane >> 4;
  v8h lo = *(const v8h*)(row + 8 * g);
  v8h hi = *(const v8h*)(row + 16 + 8 * g);
  v16h b;
#pragma unroll
  for (int i = 0; i < 8; ++i) { b[i] = lo[i]; b[8 + i] = hi[i]; }
  return b;
}

// Low 32 bits of a generic pointer to __shared__ == wave-relative LDS offset
// (ISA 10.2: LDS aperture access truncates to addr[31:0]).
__device__ __forceinline__ unsigned lds_off(const void* p) {
  return (unsigned)(unsigned long long)p;
}

#if USE_ASYNC
// GLOBAL_LOAD_ASYNC_TO_LDS_B128: LDS[vdst] = MEM[vaddr], tracked by ASYNCcnt.
__device__ __forceinline__ void async_b128(unsigned lds, const _Float16* src) {
  asm volatile("global_load_async_to_lds_b128 %0, %1, off"
               :: "v"(lds), "v"((unsigned long long)src)
               : "memory");
}
__device__ __forceinline__ void wait_async0() {
  asm volatile("s_wait_asynccnt 0" ::: "memory");
}
#endif

struct GemmP {
  const _Float16* A;  long long sAz;  int lda;
  const _Float16* W;  long long sWz;  int ldw;
  int M, Nout, K;
  float*    h;       // f32 residual / dense output
  float*    outf;    // raw f32 output (attention scores)
  _Float16* out16;   // f16 output
  const float* bias;
  const float* bias2;
  _Float16* q16; _Float16* k16; _Float16* vT16;
  float scale;
};

// EPI: 0=patch-embed scatter  1=qkv split  2=raw S  3=attn-O scatter
//      4=residual add into h  5=GELU(erf) -> f16
template <int EPI>
__global__ __launch_bounds__(256)
void gemm_wmma(GemmP p) {
  __shared__ _Float16 As[2 * BM * LDT];   //  10 KB
  __shared__ _Float16 Bs[2 * BN * LDT];   //  40 KB

  const int tid  = threadIdx.x;
  const int lane = tid & 31;
  const int wid  = tid >> 5;
  const int wm   = wid >> 2;   // 0..1  (32-row slice)
  const int wn   = wid & 3;    // 0..3  (64-col slice)
  const int z     = blockIdx.z;
  const int mBase = blockIdx.y * BM;
  const int nBase = blockIdx.x * BN;

  const _Float16* Ag = p.A + (size_t)z * p.sAz;
  const _Float16* Wg = p.W + (size_t)z * p.sWz;

  // Cooperative-load mapping (loop invariant). OOB rows are CLAMPED, not
  // zeroed: WMMA outputs are per-(m,n) dot products, so garbage rows only
  // reach outputs the epilogue discards. This keeps loads branchless and
  // EXEC all-ones.
  const int arow = tid >> 2;            // 0..63
  const int acol = (tid & 3) * 8;       // 0,8,16,24 halves
  int mG = mBase + arow; if (mG > p.M - 1) mG = p.M - 1;
  const _Float16* aSrc = Ag + (size_t)mG * p.lda + acol;

  int nG = nBase + tid;  if (nG > p.Nout - 1) nG = p.Nout - 1;
  const _Float16* bSrc = Wg + (size_t)nG * p.ldw;   // one 64B row per thread

  const unsigned aDst0 = lds_off(As + arow * LDT + acol);
  const unsigned bDst0 = lds_off(Bs + tid * LDT);
  const unsigned aBuf = BM * LDT * 2;   // bytes per A buffer
  const unsigned bBuf = BN * LDT * 2;   // bytes per B buffer

  v8f acc[2][4];
#pragma unroll
  for (int i = 0; i < 2; ++i)
#pragma unroll
    for (int j = 0; j < 4; ++j)
#pragma unroll
      for (int e = 0; e < 8; ++e) acc[i][j][e] = 0.0f;

  const int nk = p.K / BK;

#if USE_ASYNC
  // Prologue: DMA tile 0 into buffer 0.
  async_b128(aDst0, aSrc);
#pragma unroll
  for (int c = 0; c < 4; ++c) async_b128(bDst0 + c * 16, bSrc + c * 8);

  for (int kt = 0; kt < nk; ++kt) {
    const int cur = kt & 1;
    wait_async0();                      // my DMA into buf[cur] landed
    __syncthreads();                    // everyone's landed; prev reads done
    if (kt + 1 < nk) {                  // DMA next tile into buf[cur^1]
      const _Float16* aS = aSrc + (size_t)(kt + 1) * BK;
      const _Float16* bS = bSrc + (size_t)(kt + 1) * BK;
      const unsigned aD = aDst0 + (cur ^ 1) * aBuf;
      const unsigned bD = bDst0 + (cur ^ 1) * bBuf;
      async_b128(aD, aS);
#pragma unroll
      for (int c = 0; c < 4; ++c) async_b128(bD + c * 16, bS + c * 8);
    }
    const _Float16* At = As + cur * (BM * LDT);
    const _Float16* Bt = Bs + cur * (BN * LDT);
    v16h a0 = fragA(At, wm * 32,      lane);
    v16h a1 = fragA(At, wm * 32 + 16, lane);
#pragma unroll
    for (int j = 0; j < 4; ++j) {
      v16h bj = fragB(Bt, wn * 64 + j * 16, lane);
      acc[0][j] = __builtin_amdgcn_wmma_f32_16x16x32_f16(false, a0, false, bj,
                      (short)0, acc[0][j], false, false);
      acc[1][j] = __builtin_amdgcn_wmma_f32_16x16x32_f16(false, a1, false, bj,
                      (short)0, acc[1][j], false, false);
    }
  }
#else
  for (int kt = 0; kt < nk; ++kt) {
    __syncthreads();
    {
      uint4 v = *(const uint4*)(aSrc + (size_t)kt * BK);
      *(uint4*)(As + arow * LDT + acol) = v;
      const _Float16* s = bSrc + (size_t)kt * BK;
#pragma unroll
      for (int c = 0; c < 4; ++c)
        *(uint4*)(Bs + tid * LDT + c * 8) = *(const uint4*)(s + c * 8);
    }
    __syncthreads();
    v16h a0 = fragA(As, wm * 32,      lane);
    v16h a1 = fragA(As, wm * 32 + 16, lane);
#pragma unroll
    for (int j = 0; j < 4; ++j) {
      v16h bj = fragB(Bs, wn * 64 + j * 16, lane);
      acc[0][j] = __builtin_amdgcn_wmma_f32_16x16x32_f16(false, a0, false, bj,
                      (short)0, acc[0][j], false, false);
      acc[1][j] = __builtin_amdgcn_wmma_f32_16x16x32_f16(false, a1, false, bj,
                      (short)0, acc[1][j], false, false);
    }
  }
#endif

  const int lg  = lane >> 4;
  const int lnn = lane & 15;
#pragma unroll
  for (int i = 0; i < 2; ++i)
#pragma unroll
    for (int j = 0; j < 4; ++j)
#pragma unroll
      for (int r = 0; r < 8; ++r) {
        int m = mBase + wm * 32 + i * 16 + lg * 8 + r;
        int n = nBase + wn * 64 + j * 16 + lnn;
        if (m >= p.M || n >= p.Nout) continue;
        float val = acc[i][j][r];
        if (EPI == 0) {           // patch embed -> h rows (token = 1 + patch)
          int b = m / NPATCH, pa = m % NPATCH;
          p.h[((size_t)(b * NTOK + 1 + pa)) * DIM + n] = val + p.bias[n];
        } else if (EPI == 1) {    // qkv split
          int b = m / NTOK, t = m % NTOK;
          int sec = n / DIM, cc = n - sec * DIM;
          int hh = cc >> 6, d = cc & 63;
          float bv = (sec == 0) ? p.bias[cc] : ((sec == 2) ? p.bias2[cc] : 0.0f);
          float v = val + bv;
          size_t zz = (size_t)(b * HEADS + hh);
          if (sec == 0)      p.q16[(zz * NPAD + t) * HD + d] = (_Float16)(v * p.scale);
          else if (sec == 1) p.k16[(zz * NPAD + t) * HD + d] = (_Float16)v;
          else               p.vT16[(zz * HD + d) * NPAD + t] = (_Float16)v;
        } else if (EPI == 2) {    // raw attention scores
          p.outf[(size_t)z * NPAD * NPAD + (size_t)m * NPAD + n] = val;
        } else if (EPI == 3) {    // attn output scatter -> [B,N,DIM] f16
          if (m < NTOK) {
            int b = z / HEADS, hh = z % HEADS;
            p.out16[((size_t)(b * NTOK + m)) * DIM + hh * HD + n] = (_Float16)val;
          }
        } else if (EPI == 4) {    // residual add into h
          p.h[(size_t)m * DIM + n] += val + p.bias[n];
        } else if (EPI == 5) {    // erf-GELU -> f16
          float xx = val + p.bias[n];
          float g  = 0.5f * xx * (1.0f + erff(xx * 0.70710678118654752f));
          p.out16[(size_t)m * (size_t)p.Nout + n] = (_Float16)g;
        }
      }
}

// ---------------------------------------------------------------------------
// LayerNorm over DIM=768 (biased variance, eps=1e-5). One block per token row.
// OUTMODE 0: f16 (GEMM A staging).  OUTMODE 1: f32 (final output).
// ---------------------------------------------------------------------------
template <int OUTMODE>
__global__ __launch_bounds__(256)
void ln_kernel(const float* __restrict__ x, const float* __restrict__ w,
               const float* __restrict__ b, _Float16* __restrict__ y16,
               float* __restrict__ y32) {
  __shared__ float rs[256], rq[256];
  const int row = blockIdx.x;
  const float* xr = x + (size_t)row * DIM;
  float v[3], s = 0.0f, q = 0.0f;
#pragma unroll
  for (int it = 0; it < 3; ++it) {
    float t = xr[threadIdx.x + it * 256];
    v[it] = t; s += t; q += t * t;
  }
  rs[threadIdx.x] = s; rq[threadIdx.x] = q;
  __syncthreads();
  for (int off = 128; off > 0; off >>= 1) {
    if ((int)threadIdx.x < off) {
      rs[threadIdx.x] += rs[threadIdx.x + off];
      rq[threadIdx.x] += rq[threadIdx.x + off];
    }
    __syncthreads();
  }
  float mu  = rs[0] * (1.0f / DIM);
  float var = rq[0] * (1.0f / DIM) - mu * mu;
  float inv = rsqrtf(var + 1e-5f);
#pragma unroll
  for (int it = 0; it < 3; ++it) {
    int i = threadIdx.x + it * 256;
    float o = (v[it] - mu) * inv * w[i] + b[i];
    if (OUTMODE == 0) y16[(size_t)row * DIM + i] = (_Float16)o;
    else              y32[(size_t)row * DIM + i] = o;
  }
}

// ---------------------------------------------------------------------------
// Softmax over 197 valid cols + relative-position bias (index table recomputed
// analytically). Writes f16 probabilities IN PLACE over the f32 score row
// (P row stride = 448 halves). Pad rows/cols get zeros.
// ---------------------------------------------------------------------------
__global__ __launch_bounds__(256)
void softmax_kernel(float* __restrict__ S, const float* __restrict__ rpt) {
  __shared__ float red[256];
  const int n = blockIdx.x;           // 0..223
  const int z = blockIdx.y;           // 0..383
  const int h = z % HEADS;
  float* Srow = S + ((size_t)z * NPAD + n) * NPAD;
  _Float16* Prow = (_Float16*)Srow;
  const int t = threadIdx.x;

  if (n >= NTOK) {                    // pad row: all-zero probabilities
    if (t < NPAD) Prow[t] = (_Float16)0.0f;
    return;
  }

  float val = -3.0e38f;
  if (t < NTOK) {
    int idx;
    if (n == 0 && t == 0)      idx = NRD - 1;   // 731
    else if (n == 0)           idx = NRD - 3;   // 729
    else if (t == 0)           idx = NRD - 2;   // 730
    else {
      int pi = n - 1, pj = t - 1;
      int di = pi / GRID_P - pj / GRID_P + (GRID_P - 1);
      int dj = pi % GRID_P - pj % GRID_P + (GRID_P - 1);
      idx = di * (2 * GRID_P - 1) + dj;
    }
    val = Srow[t] + rpt[idx * HEADS + h];
  }
  red[t] = val;  __syncthreads();
  for (int off = 128; off > 0; off >>= 1) {
    if (t < off) red[t] = fmaxf(red[t], red[t + off]);
    __syncthreads();
  }
  float mx = red[0];  __syncthreads();
  float e = (t < NTOK) ? __expf(val - mx) : 0.0f;
  red[t] = e;  __syncthreads();
  for (int off = 128; off > 0; off >>= 1) {
    if (t < off) red[t] += red[t + off];
    __syncthreads();
  }
  float inv = 1.0f / red[0];
  __syncthreads();                     // all f32 reads done before f16 writes
  if (t < NPAD) Prow[t] = (_Float16)((t < NTOK) ? e * inv : 0.0f);
}

// ---------------------------------------------------------------------------
// Small helpers
// ---------------------------------------------------------------------------
__global__ void cvt_kernel(const float* __restrict__ in,
                           _Float16* __restrict__ out, size_t n) {
  size_t i = (size_t)blockIdx.x * blockDim.x + threadIdx.x;
  size_t st = (size_t)gridDim.x * blockDim.x;
  for (; i < n; i += st) out[i] = (_Float16)in[i];
}

__global__ void zero16_kernel(_Float16* __restrict__ out, size_t n) {
  size_t i = (size_t)blockIdx.x * blockDim.x + threadIdx.x;
  size_t st = (size_t)gridDim.x * blockDim.x;
  for (; i < n; i += st) out[i] = (_Float16)0.0f;
}

// Patchify x[B,3,224,224] -> f16 A matrix [B*196, 3*16*16] (c,ph,pw order).
__global__ void patchify_kernel(const float* __restrict__ x,
                                _Float16* __restrict__ Apat) {
  size_t i = (size_t)blockIdx.x * blockDim.x + threadIdx.x;
  size_t st = (size_t)gridDim.x * blockDim.x;
  const size_t total = (size_t)MPATCH * DIM;
  for (; i < total; i += st) {
    int col = (int)(i % DIM);
    size_t r = i / DIM;
    int b = (int)(r / NPATCH), pa = (int)(r % NPATCH);
    int gi = pa / GRID_P, gj = pa % GRID_P;
    int c = col >> 8, ph = (col >> 4) & 15, pw = col & 15;
    Apat[i] = (_Float16)x[(((size_t)b * 3 + c) * 224 + gi * 16 + ph) * 224 +
                          gj * 16 + pw];
  }
}

__global__ void cls_kernel(const float* __restrict__ cls, float* __restrict__ h) {
  int i = blockIdx.x * blockDim.x + threadIdx.x;
  if (i < BATCH * DIM) {
    int b = i / DIM, d = i % DIM;
    h[(size_t)b * NTOK * DIM + d] = cls[d];
  }
}

// ---------------------------------------------------------------------------
// Host launcher
// ---------------------------------------------------------------------------
extern "C" void kernel_launch(void* const* d_in, const int* in_sizes, int n_in,
                              void* d_out, int out_size, void* d_ws, size_t ws_size,
                              hipStream_t stream) {
  const float* x         = (const float*)d_in[0];
  const float* patch_w   = (const float*)d_in[1];
  const float* patch_b   = (const float*)d_in[2];
  const float* cls_token = (const float*)d_in[3];
  const float* norm1_w   = (const float*)d_in[4];
  const float* norm1_b   = (const float*)d_in[5];
  const float* qkv_w     = (const float*)d_in[6];
  const float* q_bias    = (const float*)d_in[7];
  const float* v_bias    = (const float*)d_in[8];
  const float* rpb_table = (const float*)d_in[9];
  const float* proj_w    = (const float*)d_in[10];
  const float* proj_b    = (const float*)d_in[11];
  const float* norm2_w   = (const float*)d_in[12];
  const float* norm2_b   = (const float*)d_in[13];
  const float* fc1_w     = (const float*)d_in[14];
  const float* fc1_b     = (const float*)d_in[15];
  const float* fc2_w     = (const float*)d_in[16];
  const float* fc2_b     = (const float*)d_in[17];
  const float* normf_w   = (const float*)d_in[18];
  const float* normf_b   = (const float*)d_in[19];

  // ---- workspace carve (256B aligned) ----
  char* ws = (char*)d_ws;
  size_t off = 0;
  auto alloc = [&](size_t bytes) -> char* {
    char* pp = ws + off;
    off += (bytes + 255) & ~(size_t)255;
    return pp;
  };
  _Float16* patch16 = (_Float16*)alloc((size_t)DIM * DIM * 2);
  _Float16* qkv16   = (_Float16*)alloc((size_t)DEPTH * 3 * DIM * DIM * 2);
  _Float16* proj16  = (_Float16*)alloc((size_t)DEPTH * DIM * DIM * 2);
  _Float16* fc1_16  = (_Float16*)alloc((size_t)DEPTH * FF * DIM * 2);
  _Float16* fc2_16  = (_Float16*)alloc((size_t)DEPTH * DIM * FF * 2);
  float*    h       = (float*)   alloc((size_t)MTOK * DIM * 4);
  _Float16* act16   = (_Float16*)alloc((size_t)MTOK * DIM * 2);   // LN-out / attn-out
  _Float16* a1_16   = (_Float16*)alloc((size_t)MTOK * FF * 2);    // also aliases patch A
  _Float16* q16     = (_Float16*)alloc((size_t)ZHEADS * NPAD * HD * 2);
  _Float16* k16     = (_Float16*)alloc((size_t)ZHEADS * NPAD * HD * 2);
  _Float16* vT16    = (_Float16*)alloc((size_t)ZHEADS * HD * NPAD * 2);
  float*    S       = (float*)   alloc((size_t)ZHEADS * NPAD * NPAD * 4);
  _Float16* patchA  = a1_16;    // alias: patch A matrix dead before fc1 runs

  auto gemm = [&](int epi, const GemmP& p, int Z) {
    dim3 g((p.Nout + BN - 1) / BN, (p.M + BM - 1) / BM, Z);
    switch (epi) {
      case 0: gemm_wmma<0><<<g, 256, 0, stream>>>(p); break;
      case 1: gemm_wmma<1><<<g, 256, 0, stream>>>(p); break;
      case 2: gemm_wmma<2><<<g, 256, 0, stream>>>(p); break;
      case 3: gemm_wmma<3><<<g, 256, 0, stream>>>(p); break;
      case 4: gemm_wmma<4><<<g, 256, 0, stream>>>(p); break;
      case 5: gemm_wmma<5><<<g, 256, 0, stream>>>(p); break;
    }
  };

  // ---- one-time f32->f16 weight conversion ----
  cvt_kernel<<<2048, 256, 0, stream>>>(patch_w, patch16, (size_t)DIM * DIM);
  cvt_kernel<<<4096, 256, 0, stream>>>(qkv_w,  qkv16,  (size_t)DEPTH * 3 * DIM * DIM);
  cvt_kernel<<<4096, 256, 0, stream>>>(proj_w, proj16, (size_t)DEPTH * DIM * DIM);
  cvt_kernel<<<4096, 256, 0, stream>>>(fc1_w,  fc1_16, (size_t)DEPTH * FF * DIM);
  cvt_kernel<<<4096, 256, 0, stream>>>(fc2_w,  fc2_16, (size_t)DEPTH * DIM * FF);
  // zero q/k/vT pad rows once (contiguous region; epilogues never touch pads)
  zero16_kernel<<<4096, 256, 0, stream>>>(q16, (size_t)3 * ZHEADS * NPAD * HD);

  // ---- patch embed ----
  patchify_kernel<<<4096, 256, 0, stream>>>(x, patchA);
  {
    GemmP p{};
    p.A = patchA; p.lda = DIM; p.M = MPATCH;
    p.W = patch16; p.ldw = DIM; p.Nout = DIM; p.K = DIM;
    p.h = h; p.bias = patch_b;
    gemm(0, p, 1);
  }
  cls_kernel<<<(BATCH * DIM + 255) / 256, 256, 0, stream>>>(cls_token, h);

  // ---- transformer blocks ----
  for (int l = 0; l < DEPTH; ++l) {
    const float* n1w = norm1_w + (size_t)l * DIM;
    const float* n1b = norm1_b + (size_t)l * DIM;
    const float* n2w = norm2_w + (size_t)l * DIM;
    const float* n2b = norm2_b + (size_t)l * DIM;

    ln_kernel<0><<<MTOK, 256, 0, stream>>>(h, n1w, n1b, act16, nullptr);

    {  // qkv
      GemmP p{};
      p.A = act16; p.lda = DIM; p.M = MTOK;
      p.W = qkv16 + (size_t)l * 3 * DIM * DIM; p.ldw = DIM;
      p.Nout = 3 * DIM; p.K = DIM;
      p.bias = q_bias + (size_t)l * DIM; p.bias2 = v_bias + (size_t)l * DIM;
      p.q16 = q16; p.k16 = k16; p.vT16 = vT16; p.scale = 0.125f;  // 64^-0.5
      gemm(1, p, 1);
    }
    {  // S = Q K^T  (batched over 384 heads)
      GemmP p{};
      p.A = q16; p.sAz = (long long)NPAD * HD; p.lda = HD;
      p.W = k16; p.sWz = (long long)NPAD * HD; p.ldw = HD;
      p.M = NPAD; p.Nout = NPAD; p.K = HD;
      p.outf = S;
      gemm(2, p, ZHEADS);
    }
    softmax_kernel<<<dim3(NPAD, ZHEADS), 256, 0, stream>>>(
        S, rpb_table + (size_t)l * NRD * HEADS);
    {  // O = P V  (P is f16 in place over S, row stride 448 halves)
      GemmP p{};
      p.A = (const _Float16*)S; p.sAz = (long long)NPAD * (2 * NPAD);
      p.lda = 2 * NPAD;
      p.W = vT16; p.sWz = (long long)HD * NPAD; p.ldw = NPAD;
      p.M = NPAD; p.Nout = HD; p.K = NPAD;
      p.out16 = act16;
      gemm(3, p, ZHEADS);
    }
    {  // proj + residual
      GemmP p{};
      p.A = act16; p.lda = DIM; p.M = MTOK;
      p.W = proj16 + (size_t)l * DIM * DIM; p.ldw = DIM;
      p.Nout = DIM; p.K = DIM;
      p.h = h; p.bias = proj_b + (size_t)l * DIM;
      gemm(4, p, 1);
    }

    ln_kernel<0><<<MTOK, 256, 0, stream>>>(h, n2w, n2b, act16, nullptr);

    {  // fc1 + erf-GELU
      GemmP p{};
      p.A = act16; p.lda = DIM; p.M = MTOK;
      p.W = fc1_16 + (size_t)l * FF * DIM; p.ldw = DIM;
      p.Nout = FF; p.K = DIM;
      p.bias = fc1_b + (size_t)l * FF; p.out16 = a1_16;
      gemm(5, p, 1);
    }
    {  // fc2 + residual
      GemmP p{};
      p.A = a1_16; p.lda = FF; p.M = MTOK;
      p.W = fc2_16 + (size_t)l * DIM * FF; p.ldw = FF;
      p.Nout = DIM; p.K = FF;
      p.h = h; p.bias = fc2_b + (size_t)l * DIM;
      gemm(4, p, 1);
    }
  }

  // ---- final layernorm -> d_out (f32 [32,197,768]) ----
  ln_kernel<1><<<MTOK, 256, 0, stream>>>(h, normf_w, normf_b, nullptr,
                                         (float*)d_out);
}
